// SwinTransformerBlock_76373108458051
// MI455X (gfx1250) — matrix operand order repaired and checked by reading, not compiled, not run on hardware
//
#include <hip/hip_runtime.h>
#include <hip/hip_bf16.h>
#include <math.h>

typedef __bf16 bf16_t;
typedef bf16_t v16bf __attribute__((ext_vector_type(16)));
typedef float  v8f   __attribute__((ext_vector_type(8)));
typedef unsigned short u16x8 __attribute__((ext_vector_type(8)));

// ---------- helpers ----------
__device__ __forceinline__ unsigned short f2bf(float f) {
    unsigned int u = __builtin_bit_cast(unsigned int, f);
    u += 0x7fffu + ((u >> 16) & 1u);          // round-to-nearest-even
    return (unsigned short)(u >> 16);
}

__device__ __forceinline__ v8f zero8() {
    v8f z;
#pragma unroll
    for (int i = 0; i < 8; ++i) z[i] = 0.f;
    return z;
}

union FragCvt { u16x8 h[2]; v16bf v; };

// A fragment: 16x32 bf16 tile, row-major, stride ld (elements, ld%8==0).
// Lane(half,m): two contiguous 8-elem runs at K = half*8 and 16+half*8.
__device__ __forceinline__ v16bf load_a_frag(const unsigned short* s, int ld) {
    int lane = threadIdx.x & 31;
    int half = lane >> 4, m = lane & 15;
    const unsigned short* row = s + m * ld + half * 8;
    FragCvt u;
    u.h[0] = *(const u16x8*)(row);
    u.h[1] = *(const u16x8*)(row + 16);
    return u.v;
}

// B fragment from a TRANSPOSED tile: LDS holds Bt_t[n][k] (i.e. W[out][in] layout),
// stride ld (elements, ld%8==0). Lane(half,n): two contiguous 8-elem runs at k = half*8 and 16+half*8.
__device__ __forceinline__ v16bf load_b_fragT(const unsigned short* s, int ld) {
    int lane = threadIdx.x & 31;
    int half = lane >> 4, n = lane & 15;
    const unsigned short* row = s + n * ld + half * 8;
    FragCvt u;
    u.h[0] = *(const u16x8*)(row);
    u.h[1] = *(const u16x8*)(row + 16);
    return u.v;
}

__device__ __forceinline__ v8f wmma_bf16(v16bf a, v16bf b, v8f c) {
    return __builtin_amdgcn_wmma_f32_16x16x32_bf16(false, a, false, b, (short)0, c, false, false);
}

// ---------- weight convert: fp32 -> bf16, same [out][in] layout ----------
__global__ void wconv_kernel(const float* __restrict__ W, unsigned short* __restrict__ Wb,
                             int total) {
    int idx = blockIdx.x * 256 + threadIdx.x;
    if (idx >= total) return;
    Wb[idx] = f2bf(W[idx]);
}

// ---------- LayerNorm (one wave per token), optional shift+window-partition ----------
__global__ __launch_bounds__(256) void ln_kernel(const float* __restrict__ x,
                                                 const float* __restrict__ g,
                                                 const float* __restrict__ bta,
                                                 unsigned short* __restrict__ out,
                                                 int shifted) {
    int wave = threadIdx.x >> 5, lane = threadIdx.x & 31;
    int M = blockIdx.x * 8 + wave;          // windowed-token index
    const float* src;
    if (shifted) {
        int win = M >> 6, t = M & 63;
        int b = win >> 6, widx = win & 63;
        int hh = (((widx >> 3) * 8 + (t >> 3)) + 4) & 63;   // inverse roll(-4)
        int ww = (((widx & 7) * 8 + (t & 7)) + 4) & 63;
        src = x + ((size_t)b * 4096 + hh * 64 + ww) * 192;
    } else {
        src = x + (size_t)M * 192;
    }
    float v[6], s = 0.f, sq = 0.f;
#pragma unroll
    for (int i = 0; i < 6; ++i) {
        v[i] = src[lane + i * 32];
        s += v[i]; sq += v[i] * v[i];
    }
#pragma unroll
    for (int off = 16; off; off >>= 1) {
        s  += __shfl_xor(s,  off, 32);
        sq += __shfl_xor(sq, off, 32);
    }
    float mean = s * (1.f / 192.f);
    float var  = sq * (1.f / 192.f) - mean * mean;
    float rstd = rsqrtf(var + 1e-5f);
#pragma unroll
    for (int i = 0; i < 6; ++i) {
        int c = lane + i * 32;
        out[(size_t)M * 192 + c] = f2bf((v[i] - mean) * rstd * g[c] + bta[c]);
    }
}

// ---------- generic bf16 WMMA GEMM, 128x64 block tile ----------
// B source is the weight in original [N][K] bf16 layout (tiles already "transposed").
// MODE 0: QKV  -> scatter into [3][2048][6][64][32] bf16, scale q
// MODE 1: proj -> window-reverse + roll + residual(x) -> x2 fp32
// MODE 2: fc1  -> +bias, exact GELU -> bf16
// MODE 3: fc2  -> +bias + residual(x2) -> fp32 out
template <int MODE>
__global__ __launch_bounds__(256) void gemm_bf16_kernel(
    const unsigned short* __restrict__ A,    // [Mtot x K] bf16
    const unsigned short* __restrict__ W,    // [N x K] bf16 (original layout)
    const float* __restrict__ bias,          // [N]
    const float* __restrict__ resid,         // MODE1: x, MODE3: x2
    unsigned short* __restrict__ out_bf,     // MODE0/2
    float* __restrict__ out_f,               // MODE1/3
    int K, int N) {
    constexpr int BM = 128, BN = 64, BK = 32, LDA = 40, LDB = 40;
    __shared__ unsigned short As[BM * LDA];
    __shared__ unsigned short Bs[BN * LDB];  // [n][k] layout

    int m0 = blockIdx.x * BM;
    int n0 = blockIdx.y * BN;
    int tid = threadIdx.x;
    int wave = tid >> 5, lane = tid & 31;
    int wm = wave >> 1, wn = wave & 1;

    v8f acc[2][2];
#pragma unroll
    for (int i = 0; i < 2; ++i)
#pragma unroll
        for (int j = 0; j < 2; ++j) acc[i][j] = zero8();

    for (int k0 = 0; k0 < K; k0 += BK) {
        // A tile: 128 rows x 32 cols = 512 uint4, 2 per thread
#pragma unroll
        for (int i = 0; i < 2; ++i) {
            int idx = tid + i * 256;
            int r = idx >> 2, c4 = idx & 3;   // 4 uint4 per 32-elem row
            uint4 d = *(const uint4*)(A + (size_t)(m0 + r) * K + k0 + c4 * 8);
            *(uint4*)&As[r * LDA + c4 * 8] = d;
        }
        // B tile: 64 rows (n) x 32 cols (k) = 256 uint4, 1 per thread
        {
            int r = tid >> 2, c4 = tid & 3;
            uint4 d = *(const uint4*)(W + (size_t)(n0 + r) * K + k0 + c4 * 8);
            *(uint4*)&Bs[r * LDB + c4 * 8] = d;
        }
        __syncthreads();
        v16bf af[2], bf[2];
#pragma unroll
        for (int i = 0; i < 2; ++i) af[i] = load_a_frag(&As[(wm * 32 + i * 16) * LDA], LDA);
#pragma unroll
        for (int j = 0; j < 2; ++j) bf[j] = load_b_fragT(&Bs[(wn * 32 + j * 16) * LDB], LDB);
#pragma unroll
        for (int i = 0; i < 2; ++i)
#pragma unroll
            for (int j = 0; j < 2; ++j) acc[i][j] = wmma_bf16(af[i], bf[j], acc[i][j]);
        __syncthreads();
    }

    int half = lane >> 4, nl = lane & 15;
#pragma unroll
    for (int i = 0; i < 2; ++i) {
#pragma unroll
        for (int j = 0; j < 2; ++j) {
            int tm = m0 + wm * 32 + i * 16 + half * 8;
            int Nc = n0 + wn * 32 + j * 16 + nl;
            float bia = bias[Nc];
            if constexpr (MODE == 0) {
                int which = Nc / 192;
                int c = Nc - which * 192;
                int head = c >> 5, d = c & 31;
                float scale = (which == 0) ? 0.1767766952966369f : 1.f; // 1/sqrt(32)
                unsigned short* outp =
                    out_bf + ((size_t)which * 2048 * 6) * 2048 + (size_t)head * 2048 + d;
#pragma unroll
                for (int r = 0; r < 8; ++r) {
                    int M = tm + r;
                    float v = (acc[i][j][r] + bia) * scale;
                    int win = M >> 6, t = M & 63;
                    outp[((size_t)win * 6) * 2048 + t * 32] = f2bf(v);
                }
            } else if constexpr (MODE == 1) {
#pragma unroll
                for (int r = 0; r < 8; ++r) {
                    int M = tm + r;
                    float v = acc[i][j][r] + bia;
                    int win = M >> 6, t = M & 63;
                    int b = win >> 6, widx = win & 63;
                    int hs  = (widx >> 3) * 8 + (t >> 3);
                    int wsc = (widx & 7) * 8 + (t & 7);
                    int hh = (hs + 4) & 63, ww = (wsc + 4) & 63;
                    size_t o = ((size_t)b * 4096 + hh * 64 + ww) * 192 + Nc;
                    out_f[o] = resid[o] + v;
                }
            } else if constexpr (MODE == 2) {
#pragma unroll
                for (int r = 0; r < 8; ++r) {
                    int M = tm + r;
                    float v = acc[i][j][r] + bia;
                    v = 0.5f * v * (1.f + erff(v * 0.7071067811865475f));
                    out_bf[(size_t)M * 768 + Nc] = f2bf(v);
                }
            } else {
#pragma unroll
                for (int r = 0; r < 8; ++r) {
                    int M = tm + r;
                    size_t o = (size_t)M * 192 + Nc;
                    out_f[o] = acc[i][j][r] + bia + resid[o];
                }
            }
        }
    }
}

// ---------- fused window attention: one block per (window, head) ----------
__global__ __launch_bounds__(128) void attn_kernel(const unsigned short* __restrict__ qkv,
                                                   const float* __restrict__ rpb,
                                                   unsigned short* __restrict__ obf) {
    __shared__ unsigned short qs[64 * 40];   // q  [token][dim]  (A tile)
    __shared__ unsigned short ks[64 * 40];   // k  [token][dim]  (B tile, [n][k] layout)
    __shared__ unsigned short vts[32 * 72];  // v^T [dim][token] (B tile, [n][k] layout)
    __shared__ float          S[64 * 68];    // scores
    __shared__ unsigned short P[64 * 72];    // softmax(S) bf16 (A tile)
    __shared__ int            regsh[64];     // shift-mask region code per token

    int wh = blockIdx.x;
    int win = wh / 6, head = wh - win * 6;
    int widx = win & 63;
    int tid = threadIdx.x;

    const size_t WHICH = (size_t)2048 * 6 * 2048;
    size_t base = ((size_t)win * 6 + head) * 2048;
    const unsigned short* qg = qkv + base;
    const unsigned short* kg = qkv + WHICH + base;
    const unsigned short* vg = qkv + 2 * WHICH + base;

    // q,k direct copies: 256 uint4 each, 2 per thread
#pragma unroll
    for (int i = 0; i < 2; ++i) {
        int idx = tid + i * 128;
        int r = idx >> 2, c4 = idx & 3;
        *(uint4*)&qs[r * 40 + c4 * 8] = ((const uint4*)qg)[idx];
        *(uint4*)&ks[r * 40 + c4 * 8] = ((const uint4*)kg)[idx];
    }
    // v transpose: 1024 uints, 8 per thread; uint = 2 consecutive dims of one token
#pragma unroll
    for (int i = 0; i < 8; ++i) {
        int idx = tid + i * 128;
        int t = idx >> 4, d2 = idx & 15;
        unsigned int u = ((const unsigned int*)vg)[idx];
        vts[(2 * d2) * 72 + t]     = (unsigned short)(u & 0xffff);
        vts[(2 * d2 + 1) * 72 + t] = (unsigned short)(u >> 16);
    }
    // region codes for the shifted-window mask (token -> region id)
    if (tid < 64) {
        int ny = tid >> 3, nx = tid & 7;
        int hsn = (widx >> 3) * 8 + ny, wsn = (widx & 7) * 8 + nx;
        regsh[tid] = (hsn < 56 ? 0 : (hsn < 60 ? 1 : 2)) * 3 +
                     (wsn < 56 ? 0 : (wsn < 60 ? 1 : 2));
    }
    __syncthreads();

    int wave = tid >> 5, lane = tid & 31;
    int half = lane >> 4, nl = lane & 15;
    int m0 = wave * 16;

    // S = q @ k^T  (M=64,N=64,K=32): one WMMA per 16x16 tile
    v16bf aq = load_a_frag(&qs[m0 * 40], 40);
#pragma unroll
    for (int nt = 0; nt < 4; ++nt) {
        v16bf bk = load_b_fragT(&ks[(nt * 16) * 40], 40);
        v8f d = wmma_bf16(aq, bk, zero8());
#pragma unroll
        for (int r = 0; r < 8; ++r)
            S[(m0 + half * 8 + r) * 68 + nt * 16 + nl] = d[r];
    }
    __syncthreads();

    // bias + mask + softmax: 2 threads per row (32-col strips, pair-combine via shfl)
    {
        int m = tid >> 1;
        int cbase = (tid & 1) * 32;
        int my = m >> 3, mx = m & 7;
        int rm = regsh[m];
        float mval = -1e30f;
        for (int nn = 0; nn < 32; ++nn) {
            int n = cbase + nn;
            int ny = n >> 3, nx = n & 7;
            float bias = rpb[((my - ny + 7) * 15 + (mx - nx + 7)) * 6 + head];
            float vv = S[m * 68 + n] + bias + (rm == regsh[n] ? 0.f : -100.f);
            S[m * 68 + n] = vv;
            mval = fmaxf(mval, vv);
        }
        mval = fmaxf(mval, __shfl_xor(mval, 1, 32));
        float sum = 0.f;
        for (int nn = 0; nn < 32; ++nn) {
            int n = cbase + nn;
            float e = __expf(S[m * 68 + n] - mval);
            S[m * 68 + n] = e;
            sum += e;
        }
        sum += __shfl_xor(sum, 1, 32);
        float inv = 1.f / sum;
        for (int nn = 0; nn < 32; ++nn) {
            int n = cbase + nn;
            P[m * 72 + n] = f2bf(S[m * 68 + n] * inv);
        }
    }
    __syncthreads();

    // O = P @ V  (M=64,N=32,K=64)
#pragma unroll
    for (int nt = 0; nt < 2; ++nt) {
        v8f acc = zero8();
#pragma unroll
        for (int ks2 = 0; ks2 < 2; ++ks2) {
            v16bf ap = load_a_frag(&P[m0 * 72 + ks2 * 32], 72);
            v16bf bv = load_b_fragT(&vts[(nt * 16) * 72 + ks2 * 32], 72);
            acc = wmma_bf16(ap, bv, acc);
        }
#pragma unroll
        for (int r = 0; r < 8; ++r) {
            int t = m0 + half * 8 + r;
            obf[((size_t)win * 64 + t) * 192 + head * 32 + nt * 16 + nl] = f2bf(acc[r]);
        }
    }
}

// ---------- host launcher ----------
extern "C" void kernel_launch(void* const* d_in, const int* in_sizes, int n_in,
                              void* d_out, int out_size, void* d_ws, size_t ws_size,
                              hipStream_t stream) {
    const float* x      = (const float*)d_in[0];
    const float* n1g    = (const float*)d_in[3];
    const float* n1b    = (const float*)d_in[4];
    const float* qkv_w  = (const float*)d_in[5];
    const float* qkv_b  = (const float*)d_in[6];
    const float* rpb    = (const float*)d_in[7];
    const float* proj_w = (const float*)d_in[8];
    const float* proj_b = (const float*)d_in[9];
    const float* n2g    = (const float*)d_in[10];
    const float* n2b    = (const float*)d_in[11];
    const float* fc1_w  = (const float*)d_in[12];
    const float* fc1_b  = (const float*)d_in[13];
    const float* fc2_w  = (const float*)d_in[14];
    const float* fc2_b  = (const float*)d_in[15];
    float* out = (float*)d_out;
    char* ws = (char*)d_ws;

    // workspace layout (bytes)
    unsigned short* wq   = (unsigned short*)(ws + 0);          //  221184  qkv_w bf16 [576][192]
    unsigned short* wp   = (unsigned short*)(ws + 221184);     //   73728  proj_w bf16 [192][192]
    unsigned short* w1   = (unsigned short*)(ws + 294912);     //  294912  fc1_w bf16 [768][192]
    unsigned short* w2   = (unsigned short*)(ws + 589824);     //  294912  fc2_w bf16 [192][768]
    unsigned short* xw   = (unsigned short*)(ws + 884736);     // 50331648 windowed LN1 out (reused as xn2)
    unsigned short* qkvb = (unsigned short*)(ws + 51216384);   // 150994944 [3][2048][6][64][32]
    unsigned short* obf  = (unsigned short*)(ws + 202211328);  // 50331648 attention out
    float*          x2   = (float*)(ws + 252542976);           // 100663296 residual-1 output
    unsigned short* hbuf = qkvb;                               // 201326592 fc1 out (reuses qkv+obf)
    unsigned short* xn2  = xw;                                 // LN2 out (reuses xw)

    // 1) weight conversion to bf16 (layout preserved: [out][in])
    wconv_kernel<<<(576 * 192 + 255) / 256, 256, 0, stream>>>(qkv_w, wq, 576 * 192);
    wconv_kernel<<<(192 * 192 + 255) / 256, 256, 0, stream>>>(proj_w, wp, 192 * 192);
    wconv_kernel<<<(768 * 192 + 255) / 256, 256, 0, stream>>>(fc1_w, w1, 768 * 192);
    wconv_kernel<<<(192 * 768 + 255) / 256, 256, 0, stream>>>(fc2_w, w2, 192 * 768);

    // 2) LN1 + shift + window partition -> bf16
    ln_kernel<<<16384, 256, 0, stream>>>(x, n1g, n1b, xw, 1);

    // 3) QKV GEMM [131072x192] x [192->576]
    gemm_bf16_kernel<0><<<dim3(1024, 9), 256, 0, stream>>>(xw, wq, qkv_b, nullptr, qkvb, nullptr, 192, 576);

    // 4) windowed attention (2048 windows x 6 heads)
    attn_kernel<<<12288, 128, 0, stream>>>(qkvb, rpb, obf);

    // 5) proj GEMM + window-reverse + roll + residual -> x2 fp32
    gemm_bf16_kernel<1><<<dim3(1024, 3), 256, 0, stream>>>(obf, wp, proj_b, x, nullptr, x2, 192, 192);

    // 6) LN2 -> bf16
    ln_kernel<<<16384, 256, 0, stream>>>(x2, n2g, n2b, xn2, 0);

    // 7) fc1 GEMM + GELU -> bf16
    gemm_bf16_kernel<2><<<dim3(1024, 12), 256, 0, stream>>>(xn2, w1, fc1_b, nullptr, hbuf, nullptr, 192, 768);

    // 8) fc2 GEMM + residual -> fp32 out
    gemm_bf16_kernel<3><<<dim3(1024, 3), 256, 0, stream>>>(hbuf, w2, fc2_b, x2, nullptr, out, 768, 192);
}
